// MultiHeadAttention_46462956208967
// MI455X (gfx1250) — matrix operand order repaired and checked by reading, not compiled, not run on hardware
//
#include <hip/hip_runtime.h>
#include <hip/hip_bf16.h>
#include <stdint.h>

#define S_LEN 2048
#define NHEADS 32
#define WIN 128

typedef __attribute__((ext_vector_type(16))) __bf16 bf16x16;
typedef __attribute__((ext_vector_type(8)))  float  f32x8;

union FragAB { bf16x16 v; uint4 q[2]; };

__device__ __forceinline__ __bf16 f2bf(float x) {
  union { float f; uint32_t u; } a; a.f = x;
  uint32_t r = a.u + 0x7FFFu + ((a.u >> 16) & 1u);
  union { uint16_t u; __bf16 h; } o; o.u = (uint16_t)(r >> 16);
  return o.h;
}

// LDS byte offset of a __shared__ pointer: on AMDGPU the shared->generic cast
// is {aperture_hi, lds_offset}, so the low 32 bits are the DS address.
__device__ __forceinline__ uint32_t lds_off(const void* p) {
  return (uint32_t)(uintptr_t)p;
}

// CDNA5 async copy: global -> LDS, 16 bytes/lane, tracked by ASYNCcnt.
__device__ __forceinline__ void async_b128(uint32_t lds, const void* gaddr) {
  asm volatile("global_load_async_to_lds_b128 %0, %1, off"
               :: "v"(lds), "v"(gaddr) : "memory");
}
template <int N>
__device__ __forceinline__ void wait_async() {
  asm volatile("s_wait_asynccnt %0" :: "n"(N) : "memory");
}

// ---------------------------------------------------------------- convert ---
__global__ void cvt_f32_bf16(const float* __restrict__ src,
                             __bf16* __restrict__ dst, int n) {
  int i = blockIdx.x * 256 + threadIdx.x;
  if (i < n) dst[i] = f2bf(src[i]);
}

// ------------------------------------------------------------------- GEMM ---
// C[M,N] = A[M,K] * B[N,K]^T (bf16, row-major, transposed-B so K is contiguous
// for both fragments). 128x128 tile, 8 waves (4M x 2N), wave = 32x64 strip =
// 2x4 f32 accum fragments, K-step 32. Double-buffered LDS filled by the async
// DMA path (global_load_async_to_lds_b128) so tile i+1 streams in while the
// matrix cores consume tile i.
// mode 0: fused RMS-norm + RoPE -> bf16 (Q/K; head width 64 == wave N strip)
// mode 1: plain bf16 store (V)
// mode 2: plain f32 store (output projection, straight to d_out)
__global__ __launch_bounds__(256)
void gemm_wmma(const __bf16* __restrict__ A, const __bf16* __restrict__ Bm,
               void* __restrict__ Cp, int K, int mode,
               const float* __restrict__ nrmw, int ldOut)
{
  __shared__ __align__(16) __bf16 As[2 * 128 * 40];
  __shared__ __align__(16) __bf16 Bs[2 * 128 * 40];

  const int tid  = threadIdx.x;
  const int lane = tid & 31, wave = tid >> 5;
  const int lm   = lane & 15, half = lane >> 4;
  const int wm   = wave & 3,  wn   = wave >> 2;
  const int m0   = blockIdx.y * 128, n0 = blockIdx.x * 128;
  const int lrow = tid >> 1, lseg = (tid & 1) * 16;

  // per-thread staging addresses (two b128 chunks per matrix per tile)
  const uint32_t lA0 = lds_off(As + lrow * 40 + lseg);
  const uint32_t lA1 = lA0 + 128 * 40 * 2;           // buffer 1, bytes
  const uint32_t lB0 = lds_off(Bs + lrow * 40 + lseg);
  const uint32_t lB1 = lB0 + 128 * 40 * 2;
  const __bf16* gA = A  + (size_t)(m0 + lrow) * K + lseg;
  const __bf16* gB = Bm + (size_t)(n0 + lrow) * K + lseg;

  f32x8 acc[2][4];
  for (int mf = 0; mf < 2; ++mf)
    for (int nf = 0; nf < 4; ++nf)
      for (int e = 0; e < 8; ++e) acc[mf][nf][e] = 0.0f;

  // prologue: start tile 0 into buffer 0
  async_b128(lA0, gA);      async_b128(lA0 + 16, gA + 8);
  async_b128(lB0, gB);      async_b128(lB0 + 16, gB + 8);

  int cur = 0;
  for (int k0 = 0; k0 < K; k0 += 32) {
    __syncthreads();  // everyone done reading the buffer we are about to fill
    if (k0 + 32 < K) {
      const uint32_t nA = cur ? lA0 : lA1, nB = cur ? lB0 : lB1;
      const __bf16* pa = gA + k0 + 32;
      const __bf16* pb = gB + k0 + 32;
      async_b128(nA, pa);   async_b128(nA + 16, pa + 8);
      async_b128(nB, pb);   async_b128(nB + 16, pb + 8);
      wait_async<4>();      // in-order: the 4 loads of the CURRENT tile done
    } else {
      wait_async<0>();
    }
    __syncthreads();

    const __bf16* Ab = As + cur * (128 * 40);
    const __bf16* Bb = Bs + cur * (128 * 40);
    FragAB af[2], bf[4];
#pragma unroll
    for (int mf = 0; mf < 2; ++mf) {
      const __bf16* base = Ab + (wm * 32 + mf * 16 + lm) * 40;
      af[mf].q[0] = *(const uint4*)(base + 16 * half);
      af[mf].q[1] = *(const uint4*)(base + 16 * half + 8);
    }
#pragma unroll
    for (int nf = 0; nf < 4; ++nf) {
      const uint4* p = (const uint4*)(Bb + (wn * 64 + nf * 16 + lm) * 40 + 16 * half);
      bf[nf].q[0] = p[0]; bf[nf].q[1] = p[1];
    }
#pragma unroll
    for (int mf = 0; mf < 2; ++mf)
#pragma unroll
      for (int nf = 0; nf < 4; ++nf)
        acc[mf][nf] = __builtin_amdgcn_wmma_f32_16x16x32_bf16(
            false, af[mf].v, false, bf[nf].v, (short)0, acc[mf][nf], false, false);
    cur ^= 1;
  }

  // ------------------------------------------------------------- epilogue --
  for (int mf = 0; mf < 2; ++mf) {
    const int rowbase = m0 + wm * 32 + mf * 16 + 8 * half;  // + r
    const int colbase = n0 + wn * 64;                       // + nf*16 + lm
    if (mode == 0) {
      float ssq[8];
#pragma unroll
      for (int r = 0; r < 8; ++r) {
        float s = 0.0f;
#pragma unroll
        for (int nf = 0; nf < 4; ++nf) { float v = acc[mf][nf][r]; s += v * v; }
        s += __shfl_xor(s, 1, 32); s += __shfl_xor(s, 2, 32);
        s += __shfl_xor(s, 4, 32); s += __shfl_xor(s, 8, 32);
        ssq[r] = rsqrtf(s * (1.0f / 64.0f) + 1e-6f);
      }
      float w[4], vv[4][8];
#pragma unroll
      for (int nf = 0; nf < 4; ++nf) w[nf] = nrmw[nf * 16 + lm];
#pragma unroll
      for (int nf = 0; nf < 4; ++nf)
#pragma unroll
        for (int r = 0; r < 8; ++r) vv[nf][r] = acc[mf][nf][r] * ssq[r] * w[nf];
      const float invf0 = __expf(-0.28782314f * (float)lm);        // ln(1e4)/32
      const float invf1 = __expf(-0.28782314f * (float)(16 + lm));
#pragma unroll
      for (int r = 0; r < 8; ++r) {
        float pos = (float)((rowbase + r) & (S_LEN - 1));
        float s0, c0, s1, c1;
        __sincosf(pos * invf0, &s0, &c0);
        __sincosf(pos * invf1, &s1, &c1);
        float a0 = vv[0][r], a1 = vv[1][r], a2 = vv[2][r], a3 = vv[3][r];
        vv[0][r] = a0 * c0 - a2 * s0;  vv[2][r] = a2 * c0 + a0 * s0;
        vv[1][r] = a1 * c1 - a3 * s1;  vv[3][r] = a3 * c1 + a1 * s1;
      }
      __bf16* dst = (__bf16*)Cp;
#pragma unroll
      for (int nf = 0; nf < 4; ++nf)
#pragma unroll
        for (int r = 0; r < 8; ++r)
          dst[(size_t)(rowbase + r) * ldOut + colbase + nf * 16 + lm] = f2bf(vv[nf][r]);
    } else if (mode == 1) {
      __bf16* dst = (__bf16*)Cp;
#pragma unroll
      for (int nf = 0; nf < 4; ++nf)
#pragma unroll
        for (int r = 0; r < 8; ++r)
          dst[(size_t)(rowbase + r) * ldOut + colbase + nf * 16 + lm] = f2bf(acc[mf][nf][r]);
    } else {
      float* dst = (float*)Cp;
#pragma unroll
      for (int nf = 0; nf < 4; ++nf)
#pragma unroll
        for (int r = 0; r < 8; ++r)
          dst[(size_t)(rowbase + r) * ldOut + colbase + nf * 16 + lm] = acc[mf][nf][r];
    }
  }
}

// -------------------------------------------------- sliding-window attention
// grid (S/64, NHEADS, B), 128 threads (4 waves x 16 queries). Window 128 means
// a 64-query block touches exactly the 3 aligned key tiles {q0-128,q0-64,q0}.
// Online softmax initialized with the sink logit: m = sinks[h], l = 1.
__global__ __launch_bounds__(128)
void attn_swa(const __bf16* __restrict__ Q, const __bf16* __restrict__ Kc,
              const __bf16* __restrict__ Vc, const int* __restrict__ amask,
              const float* __restrict__ sinks, __bf16* __restrict__ Out)
{
  __shared__ __align__(16) __bf16 Ks[64 * 72];      // [kpos][d]
  __shared__ __align__(16) __bf16 Vt[64 * 72];      // [d][kpos] (transposed)
  __shared__ __align__(16) __bf16 Ps[4 * 16 * 72];  // per-wave P transpose buf

  const int tid  = threadIdx.x;
  const int lane = tid & 31, wave = tid >> 5;
  const int lm   = lane & 15, half = lane >> 4;
  const int qb   = blockIdx.x * 64;
  const int h    = blockIdx.y, b = blockIdx.z;
  const int hkv  = h >> 2;                          // 32 heads -> 8 KV heads

  FragAB qf[2];
  {
    const __bf16* base = Q + (size_t)(b * S_LEN + qb + wave * 16 + lm) * 2048 + h * 64;
#pragma unroll
    for (int kd = 0; kd < 2; ++kd) {
      qf[kd].q[0] = *(const uint4*)(base + kd * 32 + 16 * half);
      qf[kd].q[1] = *(const uint4*)(base + kd * 32 + 16 * half + 8);
    }
  }

  float m_r[8], l_r[8];
  f32x8 o[4];
  const float sk = sinks[h];
#pragma unroll
  for (int r = 0; r < 8; ++r) { m_r[r] = sk; l_r[r] = 1.0f; }
#pragma unroll
  for (int nf = 0; nf < 4; ++nf)
    for (int e = 0; e < 8; ++e) o[nf][e] = 0.0f;

  const int ldr = tid >> 1, ldc = (tid & 1) * 32;
  const uint32_t lK = lds_off(Ks + ldr * 72 + ldc);

  for (int t = 0; t < 3; ++t) {
    const int kt = qb - 128 + t * 64;
    if (kt < 0) continue;                           // block-uniform
    { // K tile via the async DMA path; V tile transposed through VGPRs
      const __bf16* gk = Kc + (size_t)(b * S_LEN + kt + ldr) * 512 + hkv * 64 + ldc;
      async_b128(lK,      gk);      async_b128(lK + 16, gk + 8);
      async_b128(lK + 32, gk + 16); async_b128(lK + 48, gk + 24);
      union { uint4 q[4]; __bf16 h[32]; } tmp;
      const uint4* gv = (const uint4*)(Vc + (size_t)(b * S_LEN + kt + ldr) * 512 + hkv * 64 + ldc);
      tmp.q[0] = gv[0]; tmp.q[1] = gv[1]; tmp.q[2] = gv[2]; tmp.q[3] = gv[3];
#pragma unroll
      for (int j = 0; j < 32; ++j) Vt[(ldc + j) * 72 + ldr] = tmp.h[j];
      wait_async<0>();
    }
    __syncthreads();

    // S = Q K^T (16 x 64 per wave)
    f32x8 sacc[4];
#pragma unroll
    for (int nf = 0; nf < 4; ++nf)
      for (int e = 0; e < 8; ++e) sacc[nf][e] = 0.0f;
#pragma unroll
    for (int kd = 0; kd < 2; ++kd)
#pragma unroll
      for (int nf = 0; nf < 4; ++nf) {
        FragAB kb;
        const uint4* p = (const uint4*)(Ks + (nf * 16 + lm) * 72 + kd * 32 + 16 * half);
        kb.q[0] = p[0]; kb.q[1] = p[1];
        sacc[nf] = __builtin_amdgcn_wmma_f32_16x16x32_bf16(
            false, qf[kd].v, false, kb.v, (short)0, sacc[nf], false, false);
      }

    // mask + online softmax (stats per query row, replicated over 16 lanes)
    float rowmax[8];
#pragma unroll
    for (int r = 0; r < 8; ++r) rowmax[r] = -3.0e38f;
#pragma unroll
    for (int nf = 0; nf < 4; ++nf) {
      const int kpos = kt + nf * 16 + lm;
      const bool kval = amask[b * S_LEN + kpos] != 0;
#pragma unroll
      for (int r = 0; r < 8; ++r) {
        const int qpos = qb + wave * 16 + r + 8 * half;
        float lg = sacc[nf][r] * 0.125f;            // d^-0.5
        const bool ok = kval && (kpos <= qpos) && (kpos >= qpos - (WIN - 1));
        lg = ok ? lg : -3.0e38f;
        sacc[nf][r] = lg;
        rowmax[r] = fmaxf(rowmax[r], lg);
      }
    }
#pragma unroll
    for (int r = 0; r < 8; ++r) {
      float v = rowmax[r];
      v = fmaxf(v, __shfl_xor(v, 1, 32)); v = fmaxf(v, __shfl_xor(v, 2, 32));
      v = fmaxf(v, __shfl_xor(v, 4, 32)); v = fmaxf(v, __shfl_xor(v, 8, 32));
      const float mnew  = fmaxf(m_r[r], v);
      const float alpha = __expf(m_r[r] - mnew);
      m_r[r] = mnew; l_r[r] *= alpha;
#pragma unroll
      for (int nf = 0; nf < 4; ++nf) o[nf][r] *= alpha;
    }
    float rs[8];
#pragma unroll
    for (int r = 0; r < 8; ++r) rs[r] = 0.0f;
#pragma unroll
    for (int nf = 0; nf < 4; ++nf)
#pragma unroll
      for (int r = 0; r < 8; ++r) {
        const float p = __expf(sacc[nf][r] - m_r[r]);
        rs[r] += p;
        Ps[(wave * 16 + r + 8 * half) * 72 + nf * 16 + lm] = f2bf(p);
      }
#pragma unroll
    for (int r = 0; r < 8; ++r) {
      float v = rs[r];
      v += __shfl_xor(v, 1, 32); v += __shfl_xor(v, 2, 32);
      v += __shfl_xor(v, 4, 32); v += __shfl_xor(v, 8, 32);
      l_r[r] += v;
    }

    // O += P V   (P re-read from LDS in A-fragment layout; same-wave DS order)
#pragma unroll
    for (int kd = 0; kd < 2; ++kd) {
      FragAB pf;
      const __bf16* pb = Ps + (wave * 16 + lm) * 72 + kd * 32;
      pf.q[0] = *(const uint4*)(pb + 16 * half);
      pf.q[1] = *(const uint4*)(pb + 16 * half + 8);
#pragma unroll
      for (int nf = 0; nf < 4; ++nf) {
        FragAB vb;
        const uint4* p = (const uint4*)(Vt + (nf * 16 + lm) * 72 + kd * 32 + 16 * half);
        vb.q[0] = p[0]; vb.q[1] = p[1];
        o[nf] = __builtin_amdgcn_wmma_f32_16x16x32_bf16(
            false, pf.v, false, vb.v, (short)0, o[nf], false, false);
      }
    }
    __syncthreads();
  }

#pragma unroll
  for (int nf = 0; nf < 4; ++nf)
#pragma unroll
    for (int r = 0; r < 8; ++r) {
      const int qpos = qb + wave * 16 + r + 8 * half;
      Out[(size_t)(b * S_LEN + qpos) * 2048 + h * 64 + nf * 16 + lm] =
          f2bf(o[nf][r] / l_r[r]);
    }
}

// ------------------------------------------------------------------ launch --
extern "C" void kernel_launch(void* const* d_in, const int* in_sizes, int n_in,
                              void* d_out, int out_size, void* d_ws, size_t ws_size,
                              hipStream_t stream) {
  const float* x     = (const float*)d_in[0];
  const int*   amask = (const int*)  d_in[1];
  const float* wq    = (const float*)d_in[2];
  const float* wk    = (const float*)d_in[3];
  const float* wv    = (const float*)d_in[4];
  const float* wo    = (const float*)d_in[5];
  const float* qn    = (const float*)d_in[6];
  const float* kn    = (const float*)d_in[7];
  const float* snk   = (const float*)d_in[8];

  char* ws = (char*)d_ws;
  __bf16* xbf = (__bf16*)(ws);             // 4096x2048
  __bf16* wqb = (__bf16*)(ws + 16777216);  // 2048x2048
  __bf16* wkb = (__bf16*)(ws + 25165824);  //  512x2048
  __bf16* wvb = (__bf16*)(ws + 27262976);  //  512x2048
  __bf16* wob = (__bf16*)(ws + 29360128);  // 2048x2048
  __bf16* qbf = (__bf16*)(ws + 37748736);  // 4096x2048
  __bf16* kbf = (__bf16*)(ws + 54525952);  // 4096x512
  __bf16* vbf = (__bf16*)(ws + 58720256);  // 4096x512
  __bf16* abf = (__bf16*)(ws + 62914560);  // 4096x2048  (total ~76 MiB)

  cvt_f32_bf16<<<(8388608 + 255) / 256, 256, 0, stream>>>(x,  xbf, 8388608);
  cvt_f32_bf16<<<(4194304 + 255) / 256, 256, 0, stream>>>(wq, wqb, 4194304);
  cvt_f32_bf16<<<(1048576 + 255) / 256, 256, 0, stream>>>(wk, wkb, 1048576);
  cvt_f32_bf16<<<(1048576 + 255) / 256, 256, 0, stream>>>(wv, wvb, 1048576);
  cvt_f32_bf16<<<(4194304 + 255) / 256, 256, 0, stream>>>(wo, wob, 4194304);

  // Q/K projections with fused RMS-norm + RoPE; V plain.
  gemm_wmma<<<dim3(16, 32), 256, 0, stream>>>(xbf, wqb, qbf, 2048, 0, qn, 2048);
  gemm_wmma<<<dim3( 4, 32), 256, 0, stream>>>(xbf, wkb, kbf, 2048, 0, kn,  512);
  gemm_wmma<<<dim3( 4, 32), 256, 0, stream>>>(xbf, wvb, vbf, 2048, 1, nullptr, 512);

  attn_swa<<<dim3(32, NHEADS, 2), 128, 0, stream>>>(qbf, kbf, vbf, amask, snk, abf);

  // Output projection straight to f32 d_out.
  gemm_wmma<<<dim3(16, 32), 256, 0, stream>>>(abf, wob, d_out, 2048, 2, nullptr, 2048);
}
